// Block_14465449853191
// MI455X (gfx1250) — matrix operand order repaired and checked by reading, not compiled, not run on hardware
//
#include <hip/hip_runtime.h>
#include <hip/hip_bf16.h>

#define DEVI __device__ __forceinline__

typedef _Float16 v16h __attribute__((ext_vector_type(16)));
typedef _Float16 h8   __attribute__((ext_vector_type(8)));
typedef float    v8f  __attribute__((ext_vector_type(8)));

static constexpr int B_  = 2, T_ = 2048, C_ = 1024, H_ = 16, KH_ = 4, HD_ = 64;
static constexpr int E_  = 8, I_ = 1024, N_ = B_ * T_;
static constexpr float EPS_ = 1.1920929e-07f;

DEVI v16h cmb(h8 lo, h8 hi) {
  v16h r;
#pragma unroll
  for (int i = 0; i < 8; i++) { r[i] = lo[i]; r[i + 8] = hi[i]; }
  return r;
}

DEVI v8f wmma16(v16h a, v16h b, v8f c) {
  return __builtin_amdgcn_wmma_f32_16x16x32_f16(false, a, false, b, (short)0, c, false, false);
}

DEVI float rsum32(float x) {
#pragma unroll
  for (int m = 1; m < 32; m <<= 1) x += __shfl_xor(x, m, 32);
  return x;
}
DEVI float rsum16(float x) {
#pragma unroll
  for (int m = 1; m < 16; m <<= 1) x += __shfl_xor(x, m, 32);
  return x;
}
DEVI float rmax16(float x) {
#pragma unroll
  for (int m = 1; m < 16; m <<= 1) x = fmaxf(x, __shfl_xor(x, m, 32));
  return x;
}

// async global -> LDS copy of 16 bytes per lane (CDNA5 TDM-lite path, ASYNCcnt)
DEVI void async_copy_b128(unsigned ldsAddr, const void* gaddr) {
  asm volatile("global_load_async_to_lds_b128 %0, %1, off"
               :: "v"(ldsAddr), "v"(gaddr) : "memory");
}
DEVI void wait_async0() {
  asm volatile("s_wait_asynccnt 0x0" ::: "memory");
}

// ---------------------------------------------------------------- utilities
__global__ void zero_i32(int* p, int n) {
  int i = blockIdx.x * 256 + threadIdx.x;
  if (i < n) p[i] = 0;
}

__global__ void cvt_k(const float* __restrict__ s, _Float16* __restrict__ d, long n) {
  long i = ((long)blockIdx.x * 256 + threadIdx.x) * 4;
  if (i >= n) return;
  float4 v = *(const float4*)(s + i);
  d[i] = (_Float16)v.x; d[i + 1] = (_Float16)v.y;
  d[i + 2] = (_Float16)v.z; d[i + 3] = (_Float16)v.w;
}

// rmsnorm over rows of length 1024; writes f32 + f16 copies
__global__ __launch_bounds__(256) void rmsnorm_k(const float* __restrict__ x,
                                                 float* __restrict__ of,
                                                 _Float16* __restrict__ oh) {
  __shared__ float red[8];
  long rb = (long)blockIdx.x * 1024;
  float v[4]; float ss = 0.f;
#pragma unroll
  for (int i = 0; i < 4; i++) { v[i] = x[rb + threadIdx.x + 256 * i]; ss += v[i] * v[i]; }
  ss = rsum32(ss);
  int wv = threadIdx.x >> 5, lane = threadIdx.x & 31;
  if (lane == 0) red[wv] = ss;
  __syncthreads();
  float tot = 0.f;
#pragma unroll
  for (int i = 0; i < 8; i++) tot += red[i];
  float inv = rsqrtf(tot / 1024.f + EPS_);
#pragma unroll
  for (int i = 0; i < 4; i++) {
    float y = v[i] * inv;
    of[rb + threadIdx.x + 256 * i] = y;
    oh[rb + threadIdx.x + 256 * i] = (_Float16)y;
  }
}

// ------------------------------------------------ generic WMMA GEMM engine
// D[M,N] = epilogue( A[M,K] * Bw[N,K]^T ), f16 in, fp32 accumulate.
// Double-buffered async global->LDS staging (ASYNCcnt), gather resolved
// once outside the K loop. Options: batch over blockIdx.z, gathered A rows,
// row-count limit, residual add, relu^2, per-row scale, slot scatter,
// fp32 and/or f16 outputs.
__global__ __launch_bounds__(256) void gemm_wmma(
    const _Float16* __restrict__ A, long aBatch,
    const _Float16* __restrict__ Bw, long bBatch,
    int M, int N, int K,
    const int* __restrict__ mcount,
    const int* __restrict__ gather, long listBatch,
    float* __restrict__ outF32, long outF32Batch,
    _Float16* __restrict__ outF16, long outF16Batch,
    const float* __restrict__ res,
    const float* __restrict__ rowScale,
    const int* __restrict__ outRowIdx,
    const int* __restrict__ outSlot, long slotStride,
    int outPitch, int op) {
  const int z = blockIdx.z;
  const _Float16* Ab = A + (long)z * aBatch;
  const _Float16* Bb = Bw + (long)z * bBatch;
  const int cntv = mcount ? mcount[z] : M;
  const int rowBase = blockIdx.y * 128;
  if (rowBase >= cntv) return;
  const int colBase = blockIdx.x * 128;

  __shared__ __align__(16) _Float16 As[2][128][48];
  __shared__ __align__(16) _Float16 Bs[2][128][48];
  constexpr unsigned BUFB = 128u * 48u * 2u;  // bytes per buffer

  const int tid = threadIdx.x;
  const int wave = tid >> 5, lane = tid & 31, l16 = lane & 15, hf = lane >> 4;
  const int wm = wave >> 1, wn = wave & 1;

  // resolve per-thread staging sources ONCE (K-invariant, incl. gather)
  const _Float16* ag[2];
  const _Float16* bg[2];
  unsigned ldsA0[2], ldsB0[2];
#pragma unroll
  for (int p = 0; p < 2; p++) {
    int c = tid + p * 256;
    int r = c >> 2, cc = (c & 3) << 3;
    int posr = rowBase + r;
    long srcRow;
    if (gather) srcRow = (posr < cntv) ? gather[(long)z * listBatch + posr] : 0;
    else        srcRow = posr;
    ag[p] = Ab + srcRow * (long)K + cc;
    bg[p] = Bb + (long)(colBase + r) * K + cc;
    ldsA0[p] = (unsigned)(uintptr_t)&As[0][r][cc];
    ldsB0[p] = (unsigned)(uintptr_t)&Bs[0][r][cc];
  }

  auto issue = [&](int buf, int kt) {
#pragma unroll
    for (int p = 0; p < 2; p++) {
      async_copy_b128(ldsA0[p] + (unsigned)buf * BUFB, ag[p] + kt);
      async_copy_b128(ldsB0[p] + (unsigned)buf * BUFB, bg[p] + kt);
    }
  };

  v8f acc[2][4] = {};

  issue(0, 0);
  int cur = 0;
  for (int kt = 0; kt < K; kt += 32) {
    wait_async0();
    __syncthreads();
    if (kt + 32 < K) issue(cur ^ 1, kt + 32);

    const _Float16(*Ac)[48] = As[cur];
    const _Float16(*Bc)[48] = Bs[cur];
    v16h af[2], bf[4];
#pragma unroll
    for (int mi = 0; mi < 2; mi++) {
      int rr = wm * 32 + mi * 16 + l16, c0 = hf * 8;
      af[mi] = cmb(*(const h8*)&Ac[rr][c0], *(const h8*)&Ac[rr][c0 + 16]);
    }
#pragma unroll
    for (int ni = 0; ni < 4; ni++) {
      int rr = wn * 64 + ni * 16 + l16, c0 = hf * 8;
      bf[ni] = cmb(*(const h8*)&Bc[rr][c0], *(const h8*)&Bc[rr][c0 + 16]);
    }
#pragma unroll
    for (int mi = 0; mi < 2; mi++)
#pragma unroll
      for (int ni = 0; ni < 4; ni++)
        acc[mi][ni] = wmma16(af[mi], bf[ni], acc[mi][ni]);
    cur ^= 1;
  }

#pragma unroll
  for (int mi = 0; mi < 2; mi++) {
#pragma unroll
    for (int vr = 0; vr < 8; vr++) {
      int pos = rowBase + wm * 32 + mi * 16 + vr + hf * 8;
      if (pos >= cntv) continue;
      float scl = rowScale ? rowScale[(long)z * listBatch + pos] : 1.f;
      long orow = outRowIdx ? (long)outRowIdx[(long)z * listBatch + pos] : (long)pos;
      long slotOff = outSlot ? (long)outSlot[(long)z * listBatch + pos] * slotStride : 0;
      long obase = orow * (long)outPitch;
#pragma unroll
      for (int ni = 0; ni < 4; ni++) {
        int col = colBase + wn * 64 + ni * 16 + l16;
        float v = acc[mi][ni][vr];
        if (op == 1) { v = fmaxf(v, 0.f); v = v * v; }
        if (res) v += res[obase + col];
        v *= scl;
        if (outF32) outF32[(long)z * outF32Batch + slotOff + obase + col] = v;
        if (outF16) outF16[(long)z * outF16Batch + obase + col] = (_Float16)v;
      }
    }
  }
}

// --------------------------------------------- attention prep (rope+norms)
__global__ __launch_bounds__(256) void qk_prep(
    const float* __restrict__ q, const float* __restrict__ k,
    const float* __restrict__ cosp, const float* __restrict__ sinp,
    _Float16* __restrict__ qh, _Float16* __restrict__ kh) {
  int u = blockIdx.x * 8 + (threadIdx.x >> 5);
  int j = threadIdx.x & 31;
  const int NQ = N_ * H_;
  float a, bb; long dstBase; int t; _Float16* dst;
  if (u < NQ) {
    int hh = u % H_; int n = u / H_;
    t = n % T_; int b = n / T_;
    const float* s = q + (long)n * 1024 + hh * 64;
    a = s[j]; bb = s[j + 32];
    dst = qh; dstBase = ((long)(b * H_ + hh) * T_ + t) * 64;
  } else {
    int v = u - NQ; int kk = v % KH_; int n = v / KH_;
    t = n % T_; int b = n / T_;
    const float* s = k + (long)n * 256 + kk * 64;
    a = s[j]; bb = s[j + 32];
    dst = kh; dstBase = ((long)(b * KH_ + kk) * T_ + t) * 64;
  }
  float c = cosp[t * 32 + j], sn = sinp[t * 32 + j];
  float o1 = a * c + bb * sn;
  float o2 = -a * sn + bb * c;
  float ss = rsum32(o1 * o1 + o2 * o2);
  float inv = rsqrtf(ss / 64.f + EPS_);
  dst[dstBase + j]      = (_Float16)(o1 * inv);
  dst[dstBase + j + 32] = (_Float16)(o2 * inv);
}

// gate + value-embedding add, store V transposed [b][kh][hd][T] (f16)
__global__ __launch_bounds__(256) void v_prep(
    const float* __restrict__ v, const float* __restrict__ ve,
    const float* __restrict__ h, const float* __restrict__ gw,
    _Float16* __restrict__ vt) {
  int u = blockIdx.x * 8 + (threadIdx.x >> 5);  // n*KH + kh
  int j = threadIdx.x & 31;
  int kk = u % KH_; int n = u / KH_;
  int t = n % T_; int b = n / T_;
  float d = rsum32(h[(long)n * 1024 + j] * gw[kk * 32 + j]);
  float gate = 2.f / (1.f + __expf(-d));
  const float* vs = v + (long)n * 256 + kk * 64;
  const float* es = ve + (long)n * 256 + kk * 64;
  float v0 = vs[j] + gate * es[j];
  float v1 = vs[j + 32] + gate * es[j + 32];
  long dbase = (long)(b * KH_ + kk) * 64;
  vt[(dbase + j) * T_ + t]      = (_Float16)v0;
  vt[(dbase + j + 32) * T_ + t] = (_Float16)v1;
}

// ------------------------------------------------- flash attention (WMMA)
__global__ __launch_bounds__(256) void attn_wmma(
    const _Float16* __restrict__ Q, const _Float16* __restrict__ Kh,
    const _Float16* __restrict__ Vt, _Float16* __restrict__ Y,
    const int* __restrict__ wptr) {
  int b = blockIdx.z, h = blockIdx.y, qb = blockIdx.x * 128;
  int wave = threadIdx.x >> 5, lane = threadIdx.x & 31;
  int l16 = lane & 15, hf = lane >> 4;
  int qw = qb + wave * 16;
  int kh = h >> 2;
  int W = *wptr;
  const _Float16* Qp = Q + (long)(b * H_ + h) * T_ * HD_;
  const _Float16* Kp = Kh + (long)(b * KH_ + kh) * T_ * HD_;
  const _Float16* Vp = Vt + (long)(b * KH_ + kh) * HD_ * T_;
  __shared__ __align__(16) _Float16 Plds[8][16][48];

  v16h qf[2];
  {
    const _Float16* qr = Qp + (long)(qw + l16) * HD_;
#pragma unroll
    for (int kc = 0; kc < 2; kc++) {
      int c0 = kc * 32 + hf * 8;
      qf[kc] = cmb(*(const h8*)(qr + c0), *(const h8*)(qr + c0 + 16));
    }
  }

  float mrow[8], lrow[8];
#pragma unroll
  for (int r = 0; r < 8; r++) { mrow[r] = -1e30f; lrow[r] = 0.f; }
  v8f oacc[4] = {};

  for (int kvt = 0; kvt <= qw + 15; kvt += 32) {
    v8f sacc[2] = {};
#pragma unroll
    for (int s = 0; s < 2; s++) {
#pragma unroll
      for (int kc = 0; kc < 2; kc++) {
        int tK = kvt + s * 16 + l16; if (tK > T_ - 1) tK = T_ - 1;
        const _Float16* kr = Kp + (long)tK * HD_ + kc * 32 + hf * 8;
        v16h kf = cmb(*(const h8*)kr, *(const h8*)(kr + 16));
        sacc[s] = wmma16(qf[kc], kf, sacc[s]);
      }
    }
    float sv[2][8];
#pragma unroll
    for (int s = 0; s < 2; s++)
#pragma unroll
      for (int vr = 0; vr < 8; vr++) {
        int row = qw + vr + hf * 8;
        int ks = kvt + s * 16 + l16;
        float vvv = sacc[s][vr] * 0.125f;
        sv[s][vr] = (ks <= row && ks > row - W) ? vvv : -1e30f;
      }
    float alpha[8];
#pragma unroll
    for (int vr = 0; vr < 8; vr++) {
      float tm = rmax16(fmaxf(sv[0][vr], sv[1][vr]));
      float mn = fmaxf(mrow[vr], tm);
      alpha[vr] = __expf(mrow[vr] - mn);
      mrow[vr] = mn;
      float p0 = __expf(sv[0][vr] - mn);
      float p1 = __expf(sv[1][vr] - mn);
      sv[0][vr] = p0; sv[1][vr] = p1;
      float ts = rsum16(p0 + p1);
      lrow[vr] = lrow[vr] * alpha[vr] + ts;
    }
#pragma unroll
    for (int s = 0; s < 2; s++)
#pragma unroll
      for (int vr = 0; vr < 8; vr++)
        Plds[wave][vr + hf * 8][s * 16 + l16] = (_Float16)sv[s][vr];
    asm volatile("s_wait_dscnt 0" ::: "memory");
    v16h pf;
    {
      int c0 = hf * 8;
      pf = cmb(*(const h8*)&Plds[wave][l16][c0], *(const h8*)&Plds[wave][l16][c0 + 16]);
    }
#pragma unroll
    for (int j = 0; j < 4; j++)
#pragma unroll
      for (int vr = 0; vr < 8; vr++) oacc[j][vr] *= alpha[vr];
#pragma unroll
    for (int j = 0; j < 4; j++) {
      int cb = kvt + hf * 8; if (cb > T_ - 16) cb = T_ - 16;
      const _Float16* vr_ = Vp + (long)(j * 16 + l16) * T_ + cb;
      v16h vfr = cmb(*(const h8*)vr_, *(const h8*)(vr_ + 16));
      oacc[j] = wmma16(pf, vfr, oacc[j]);
    }
  }

#pragma unroll
  for (int j = 0; j < 4; j++)
#pragma unroll
    for (int vr = 0; vr < 8; vr++) {
      int t = qw + vr + hf * 8;
      float val = oacc[j][vr] / lrow[vr];
      Y[(long)(b * T_ + t) * C_ + h * 64 + j * 16 + l16] = (_Float16)val;
    }
}

// -------------------------------------------------------------- MoE router
__global__ __launch_bounds__(256) void router_k(
    const float* __restrict__ h2, const float* __restrict__ rw,
    float* __restrict__ sc) {
  int u = blockIdx.x * 8 + (threadIdx.x >> 5);  // n*E + e
  int lane = threadIdx.x & 31;
  int e = u & 7; int n = u >> 3;
  const float* hr = h2 + (long)n * 1024;
  const float* wr = rw + (long)e * 1024;
  float s = 0.f;
#pragma unroll
  for (int i = 0; i < 32; i++) s += hr[lane + 32 * i] * wr[lane + 32 * i];
  s = rsum32(s);
  if (lane == 0) sc[n * 8 + e] = 1.f / (1.f + __expf(-s));
}

__global__ void topk_k(const float* __restrict__ sc, const float* __restrict__ bias,
                       int* cnt, int* idx, float* wts, int* slot) {
  int n = blockIdx.x * 256 + threadIdx.x;
  if (n >= N_) return;
  float s[8], sb[8];
#pragma unroll
  for (int e = 0; e < 8; e++) { s[e] = sc[n * 8 + e]; sb[e] = s[e] + bias[e]; }
  int i0 = 0;
#pragma unroll
  for (int e = 1; e < 8; e++) if (sb[e] > sb[i0]) i0 = e;
  int i1 = -1;
#pragma unroll
  for (int e = 0; e < 8; e++)
    if (e != i0 && (i1 < 0 || sb[e] > sb[i1])) i1 = e;
  float w0 = s[i0], w1 = s[i1];
  float nrm = w0 + w1 + 1e-20f;
  int p0 = atomicAdd(&cnt[i0], 1);
  idx[i0 * N_ + p0] = n; wts[i0 * N_ + p0] = w0 / nrm; slot[i0 * N_ + p0] = 0;
  int p1 = atomicAdd(&cnt[i1], 1);
  idx[i1 * N_ + p1] = n; wts[i1 * N_ + p1] = w1 / nrm; slot[i1 * N_ + p1] = 1;
}

__global__ void final_k(const float* __restrict__ x2, const float* __restrict__ sh,
                        const float* __restrict__ slots, float* __restrict__ out) {
  long i = (long)blockIdx.x * 256 + threadIdx.x;
  out[i] = x2[i] + sh[i] + slots[i] + slots[(long)N_ * C_ + i];
}

// ----------------------------------------------------------------- driver
extern "C" void kernel_launch(void* const* d_in, const int* in_sizes, int n_in,
                              void* d_out, int out_size, void* d_ws, size_t ws_size,
                              hipStream_t stream) {
  (void)in_sizes; (void)n_in; (void)out_size; (void)ws_size;
  const float* x    = (const float*)d_in[0];
  const float* ve   = (const float*)d_in[1];
  const float* cosp = (const float*)d_in[2];
  const float* sinp = (const float*)d_in[3];
  const float* Wq   = (const float*)d_in[4];
  const float* Wk   = (const float*)d_in[5];
  const float* Wv   = (const float*)d_in[6];
  const float* Wo   = (const float*)d_in[7];
  const float* gW   = (const float*)d_in[8];
  const float* rW   = (const float*)d_in[9];
  const float* bias = (const float*)d_in[10];
  const float* fcW  = (const float*)d_in[11];
  const float* pjW  = (const float*)d_in[12];
  const float* efcW = (const float*)d_in[13];
  const float* epjW = (const float*)d_in[14];
  const int*   wsz  = (const int*)d_in[15];

  char* base = (char*)d_ws; size_t off = 0;
  auto alloc = [&](size_t bytes) -> char* {
    char* p = base + off;
    off = (off + bytes + 255) & ~(size_t)255;
    return p;
  };

  _Float16* wqh  = (_Float16*)alloc((size_t)C_ * C_ * 2);
  _Float16* wkh  = (_Float16*)alloc((size_t)256 * C_ * 2);
  _Float16* wvh  = (_Float16*)alloc((size_t)256 * C_ * 2);
  _Float16* woh  = (_Float16*)alloc((size_t)C_ * C_ * 2);
  _Float16* wfch = (_Float16*)alloc((size_t)I_ * C_ * 2);
  _Float16* wpjh = (_Float16*)alloc((size_t)C_ * I_ * 2);
  _Float16* wefch = (_Float16*)alloc((size_t)E_ * I_ * C_ * 2);
  _Float16* wepjh = (_Float16*)alloc((size_t)E_ * C_ * I_ * 2);
  float*    hf32 = (float*)alloc((size_t)N_ * C_ * 4);
  _Float16* h16  = (_Float16*)alloc((size_t)N_ * C_ * 2);
  float*    qf   = (float*)alloc((size_t)N_ * C_ * 4);
  float*    kf   = (float*)alloc((size_t)N_ * 256 * 4);
  float*    vf   = (float*)alloc((size_t)N_ * 256 * 4);
  _Float16* qh16 = (_Float16*)alloc((size_t)B_ * H_ * T_ * HD_ * 2);
  _Float16* kh16 = (_Float16*)alloc((size_t)B_ * KH_ * T_ * HD_ * 2);
  _Float16* vt16 = (_Float16*)alloc((size_t)B_ * KH_ * T_ * HD_ * 2);
  _Float16* y16  = (_Float16*)alloc((size_t)N_ * C_ * 2);
  float*    x2   = (float*)alloc((size_t)N_ * C_ * 4);
  float*    h2f  = (float*)alloc((size_t)N_ * C_ * 4);
  _Float16* h2h  = (_Float16*)alloc((size_t)N_ * C_ * 2);
  _Float16* t1   = (_Float16*)alloc((size_t)N_ * I_ * 2);
  float*    shf  = (float*)alloc((size_t)N_ * C_ * 4);
  float*    sc   = (float*)alloc((size_t)N_ * E_ * 4);
  int*      cnt  = (int*)alloc(8 * 4);
  int*      idx  = (int*)alloc((size_t)E_ * N_ * 4);
  float*    wts  = (float*)alloc((size_t)E_ * N_ * 4);
  int*      slot = (int*)alloc((size_t)E_ * N_ * 4);
  _Float16* he   = (_Float16*)alloc((size_t)E_ * N_ * I_ * 2);
  float*    slots = (float*)alloc((size_t)2 * N_ * C_ * 4);

  zero_i32<<<1, 32, 0, stream>>>(cnt, 8);

  auto cvt = [&](const float* s, _Float16* d, long n) {
    cvt_k<<<dim3((unsigned)((n / 4 + 255) / 256)), 256, 0, stream>>>(s, d, n);
  };
  cvt(Wq, wqh, (long)C_ * C_);
  cvt(Wk, wkh, (long)256 * C_);
  cvt(Wv, wvh, (long)256 * C_);
  cvt(Wo, woh, (long)C_ * C_);
  cvt(fcW, wfch, (long)I_ * C_);
  cvt(pjW, wpjh, (long)C_ * I_);
  cvt(efcW, wefch, (long)E_ * I_ * C_);
  cvt(epjW, wepjh, (long)E_ * C_ * I_);

  // h = rmsnorm(x)
  rmsnorm_k<<<N_, 256, 0, stream>>>(x, hf32, h16);

  // q/k/v projections (f16 WMMA, fp32 out)
  gemm_wmma<<<dim3(C_ / 128, N_ / 128, 1), 256, 0, stream>>>(
      h16, 0, wqh, 0, N_, C_, C_, nullptr, nullptr, 0,
      qf, 0, nullptr, 0, nullptr, nullptr, nullptr, nullptr, 0, C_, 0);
  gemm_wmma<<<dim3(256 / 128, N_ / 128, 1), 256, 0, stream>>>(
      h16, 0, wkh, 0, N_, 256, C_, nullptr, nullptr, 0,
      kf, 0, nullptr, 0, nullptr, nullptr, nullptr, nullptr, 0, 256, 0);
  gemm_wmma<<<dim3(256 / 128, N_ / 128, 1), 256, 0, stream>>>(
      h16, 0, wvh, 0, N_, 256, C_, nullptr, nullptr, 0,
      vf, 0, nullptr, 0, nullptr, nullptr, nullptr, nullptr, 0, 256, 0);

  // rope + head rmsnorm; gate + ve; V transposed
  qk_prep<<<(N_ * (H_ + KH_)) / 8, 256, 0, stream>>>(qf, kf, cosp, sinp, qh16, kh16);
  v_prep<<<(N_ * KH_) / 8, 256, 0, stream>>>(vf, ve, hf32, gW, vt16);

  // flash attention
  attn_wmma<<<dim3(T_ / 128, H_, B_), 256, 0, stream>>>(qh16, kh16, vt16, y16, wsz);

  // x2 = x + y @ Wo^T
  gemm_wmma<<<dim3(C_ / 128, N_ / 128, 1), 256, 0, stream>>>(
      y16, 0, woh, 0, N_, C_, C_, nullptr, nullptr, 0,
      x2, 0, nullptr, 0, x, nullptr, nullptr, nullptr, 0, C_, 0);

  // h2 = rmsnorm(x2)
  rmsnorm_k<<<N_, 256, 0, stream>>>(x2, h2f, h2h);

  // shared MLP: relu^2 fused in fc epilogue
  gemm_wmma<<<dim3(I_ / 128, N_ / 128, 1), 256, 0, stream>>>(
      h2h, 0, wfch, 0, N_, I_, C_, nullptr, nullptr, 0,
      nullptr, 0, t1, 0, nullptr, nullptr, nullptr, nullptr, 0, I_, 1);
  gemm_wmma<<<dim3(C_ / 128, N_ / 128, 1), 256, 0, stream>>>(
      t1, 0, wpjh, 0, N_, C_, I_, nullptr, nullptr, 0,
      shf, 0, nullptr, 0, nullptr, nullptr, nullptr, nullptr, 0, C_, 0);

  // router + top-2 gather lists
  router_k<<<(N_ * E_) / 8, 256, 0, stream>>>(h2f, rW, sc);
  topk_k<<<(N_ + 255) / 256, 256, 0, stream>>>(sc, bias, cnt, idx, wts, slot);

  // expert fc: gathered rows, relu^2 -> f16 (batched over experts)
  gemm_wmma<<<dim3(I_ / 128, N_ / 128, E_), 256, 0, stream>>>(
      h2h, 0, wefch, (long)I_ * C_, N_, I_, C_, cnt, idx, (long)N_,
      nullptr, 0, he, (long)N_ * I_, nullptr, nullptr, nullptr, nullptr, 0, I_, 1);
  // expert proj: scale by routing weight, scatter into slot buffers
  gemm_wmma<<<dim3(C_ / 128, N_ / 128, E_), 256, 0, stream>>>(
      he, (long)N_ * I_, wepjh, (long)C_ * I_, N_, C_, I_, cnt, nullptr, (long)N_,
      slots, 0, nullptr, 0, nullptr, wts, idx, slot, (long)N_ * C_, C_, 0);

  // out = x2 + shared + routed
  final_k<<<(N_ * C_) / 256, 256, 0, stream>>>(x2, shf, slots, (float*)d_out);
}
